// AxisNet_16088947491409
// MI455X (gfx1250) — matrix operand order repaired and checked by reading, not compiled, not run on hardware
//
#include <hip/hip_runtime.h>
#include <math.h>

typedef __attribute__((ext_vector_type(16))) _Float16 v16h;
typedef __attribute__((ext_vector_type(8)))  _Float16 v8h;
typedef __attribute__((ext_vector_type(8)))  float    v8f;

#define NB    8
#define NPT   16384
#define KNBR  15
#define NPTS  (NB*NPT)

// ---- workspace layout -------------------------------------------------------
// Wbuf: f16 at ws+0  (offsets in halves; rows padded to 16-multiples, zeroed).
// Conv layers carry their folded bias in weight column `cin` (ones-channel GEMM).
#define W_SA1_0 0       // [16][32]
#define W_SA1_1 512     // [16][32]
#define W_SA2_0 1024    // [32][32]
#define W_SA2_1 2048    // [32][32]
#define W_SA3_0 3072    // [48][64]
#define W_SA3_1 6144    // [64][64]
#define W_FC1   10240   // [32][64]
#define W_FC2   12288   // [32][32]
#define W_FC3   13312   // [16][32]
#define W_TOTAL 13824
// Bbuf: f32 at ws+28672 (only FC biases used; padded, zeroed)
#define B_FC1   208
#define B_FC2   240
#define B_FC3   272
#define B_TOTAL 288
// pooled: f16 [NPTS][64] at ws+65536 (16 MB)

struct PrepParams {
  const float* W[6]; const float* b[6]; const float* ga[6]; const float* be[6];
  const float* fv[3]; const float* fg[3]; const float* fb[3];
};

// ---- weight folding ---------------------------------------------------------
// conv:  Weff = ga*W (cols 0..cin-1), col cin = ga*b+be  (bias via ones-channel)
// fc:    weight-norm  W = g*v/||v||, bias separate (no spare input channel)
__global__ void prep_kernel(PrepParams P, _Float16* __restrict__ Wbuf,
                            float* __restrict__ Bbuf)
{
  const int COUT[6] = {8,16,24,32,48,64};
  const int CIN[6]  = {3,8,19,24,35,48};
  const int CPAD[6] = {32,32,32,32,64,64};
  const int WOFF[6] = {W_SA1_0,W_SA1_1,W_SA2_0,W_SA2_1,W_SA3_0,W_SA3_1};
  int t = threadIdx.x;
  for (int i = t; i < W_TOTAL; i += 256) Wbuf[i] = (_Float16)0.0f;
  for (int i = t; i < B_TOTAL; i += 256) Bbuf[i] = 0.0f;
  __syncthreads();
  for (int l = 0; l < 6; l++) {
    int co = COUT[l], ci = CIN[l], cp = CPAD[l];
    for (int i = t; i < co*ci; i += 256) {
      int o = i / ci, c = i - o*ci;
      Wbuf[WOFF[l] + o*cp + c] = (_Float16)(P.ga[l][o] * P.W[l][i]);
    }
    for (int o = t; o < co; o += 256)   // folded bias -> ones-channel column
      Wbuf[WOFF[l] + o*cp + ci] = (_Float16)(P.ga[l][o]*P.b[l][o] + P.be[l][o]);
  }
  const int FCO[3] = {32,32,6};
  const int FCI[3] = {64,32,32};
  const int FWOFF[3] = {W_FC1,W_FC2,W_FC3};
  const int FBOFF[3] = {B_FC1,B_FC2,B_FC3};
  for (int l = 0; l < 3; l++) {
    int co = FCO[l], ci = FCI[l];
    for (int o = t; o < co; o += 256) {
      float s = 0.f;
      for (int c = 0; c < ci; c++) { float x = P.fv[l][o*ci+c]; s += x*x; }
      float sc = P.fg[l][o] / sqrtf(s);
      for (int c = 0; c < ci; c++)
        Wbuf[FWOFF[l] + o*ci + c] = (_Float16)(sc * P.fv[l][o*ci+c]);
      Bbuf[FBOFF[l] + o] = P.fb[l][o];
    }
  }
}

// ---- WMMA helpers -----------------------------------------------------------
__device__ __forceinline__ v8f wmma16(v16h a, v16h b, v8f c) {
  return __builtin_amdgcn_wmma_f32_16x16x32_f16(false, a, false, b,
                                                (short)0, c, false, false);
}

// A fragment for one 32-K step from row-major [16][KS*32] f16 activations.
// Lane layout: M = lane&15; lanes 0-15 hold K {0..15}, lanes 16-31 hold
// K {16..31} (two contiguous 8-half chunks).
template<int KS>
__device__ __forceinline__ void loadA(const _Float16* in, int lane, v16h* A) {
  int m = lane & 15, kh = lane >> 4;
  const int CINPAD = KS * 32;
#pragma unroll
  for (int s = 0; s < KS; s++) {
    const _Float16* base = in + m*CINPAD + s*32;
    v8h lo = *(const v8h*)(base + 16*kh);
    v8h hi = *(const v8h*)(base + 16*kh + 8);
#pragma unroll
    for (int e = 0; e < 8; e++) { A[s][e] = lo[e]; A[s][e+8] = hi[e]; }
  }
}

// One layer: D = relu(A*W^T [+ bias]), written row-major [16][outStride]+outOff.
// B fragment per lane is one contiguous v16h: W[n][khalf*16 .. +16].
template<int KS, int NT, bool BIAS>
__device__ __forceinline__ void conv_layer(const _Float16* in, _Float16* out,
                                           int outStride, int outOff, int coutReal,
                                           const _Float16* __restrict__ W,
                                           const float* __restrict__ bias, int lane)
{
  const int CINPAD = KS * 32;
  int n = lane & 15, kh = lane >> 4;
  v16h A[KS];
  loadA<KS>(in, lane, A);
#pragma unroll
  for (int t = 0; t < NT; t++) {
    v8f acc = {};
#pragma unroll
    for (int s = 0; s < KS; s++) {
      v16h Bf = *(const v16h*)(W + (t*16 + n)*CINPAD + s*32 + kh*16);
      acc = wmma16(A[s], Bf, acc);
    }
    float bv = BIAS ? bias[t*16 + n] : 0.f;
    if (t*16 + n < coutReal) {
#pragma unroll
      for (int v = 0; v < 8; v++) {
        float x = BIAS ? fmaxf(acc[v] + bv, 0.f) : fmaxf(acc[v], 0.f);
        out[(v + 8*kh)*outStride + outOff + t*16 + n] = (_Float16)x;
      }
    }
  }
}

// Final sa3 layer (48->64, bias folded) fused with relu + max over 16 rows.
__device__ __forceinline__ void conv_final(const _Float16* in,
                                           const _Float16* __restrict__ W,
                                           _Float16* __restrict__ pooled,
                                           int p, int lane)
{
  int n = lane & 15, kh = lane >> 4;
  v16h A[2];
  loadA<2>(in, lane, A);
#pragma unroll
  for (int t = 0; t < 4; t++) {
    v8f acc = {};
#pragma unroll
    for (int s = 0; s < 2; s++) {
      v16h Bf = *(const v16h*)(W + (t*16 + n)*64 + s*32 + kh*16);
      acc = wmma16(A[s], Bf, acc);
    }
    float mx = 0.f;                            // relu output >= 0
#pragma unroll
    for (int v = 0; v < 8; v++) mx = fmaxf(mx, fmaxf(acc[v], 0.f));
    mx = fmaxf(mx, __shfl_xor(mx, 16, 32));    // rows 0-7 vs 8-15
    if (lane < 16) pooled[(long)p*64 + t*16 + lane] = (_Float16)mx;
  }
}

// ---- kernel 1: gather + sa1/sa2/sa3 + maxpool; one wave per point -----------
__global__ __launch_bounds__(128) void conv_kernel(
    const float* __restrict__ xyz, const int* __restrict__ nbr,
    const _Float16* __restrict__ Wbuf, _Float16* __restrict__ pooled)
{
  __shared__ __align__(16) _Float16 smem[4*3072]; // per wave: A32,B32,D64,E64
  int wave = threadIdx.x >> 5, lane = threadIdx.x & 31;
  int p = blockIdx.x*4 + wave;
  _Float16* bufA = smem + wave*3072;        // [16][32]
  _Float16* bufB = bufA + 512;              // [16][32]
  _Float16* bufD = bufA + 1024;             // [16][64]
  _Float16* bufE = bufA + 2048;             // [16][64]

  { // zero all per-wave LDS (padding channels must be 0)
    uint4 z4 = make_uint4(0,0,0,0);
    uint4* z = (uint4*)bufA;
    for (int i = lane; i < 384; i += 32) z[i] = z4;
  }

  int b = p >> 14, n = p & (NPT-1);
  float x0 = 0.f, x1 = 0.f, x2 = 0.f;
  if (lane < 16) {
    // ones-channels for bias-in-GEMM (survive the buffer reuse schedule)
    const _Float16 one = (_Float16)1.0f;
    bufA[lane*32 + 3]  = one;   // sa1 L0 input (cin=3)
    bufA[lane*32 + 19] = one;   // sa2 L0 input (cin=19)
    bufB[lane*32 + 8]  = one;   // sa1 L1 input (cin=8)
    bufB[lane*32 + 24] = one;   // sa2 L1 input (cin=24)
    bufD[lane*64 + 35] = one;   // sa3 L0 input (cin=35)
    bufE[lane*64 + 48] = one;   // sa3 L1 input (cin=48)
    // gather: row 15 duplicates neighbor 0 (max-pool invariant)
    int j   = (lane < KNBR) ? lane : 0;
    int idx = nbr[((long)(b*NPT + n))*KNBR + j];
    const float* q = xyz + ((long)(b*NPT + idx))*3;
    x0 = q[0]; x1 = q[1]; x2 = q[2];
    _Float16* r = bufA + lane*32;
    r[0] = (_Float16)x0; r[1] = (_Float16)x1; r[2] = (_Float16)x2;
  }
  __syncthreads();

  conv_layer<1,1,false>(bufA, bufB, 32, 0,  8, Wbuf + W_SA1_0, nullptr, lane);
  __syncthreads();
  conv_layer<1,1,false>(bufB, bufA, 32, 3, 16, Wbuf + W_SA1_1, nullptr, lane);
  __syncthreads();                          // bufA: xyz + l1(3-18) + 1@19
  conv_layer<1,2,false>(bufA, bufB, 32, 0, 24, Wbuf + W_SA2_0, nullptr, lane);
  if (lane < 16) {                          // xyz channels for sa3 input
    _Float16* r = bufD + lane*64;
    r[0] = (_Float16)x0; r[1] = (_Float16)x1; r[2] = (_Float16)x2;
  }
  __syncthreads();
  conv_layer<1,2,false>(bufB, bufD, 64, 3, 32, Wbuf + W_SA2_1, nullptr, lane);
  __syncthreads();                          // bufD: xyz + l2(3-34) + 1@35
  conv_layer<2,3,false>(bufD, bufE, 64, 0, 48, Wbuf + W_SA3_0, nullptr, lane);
  __syncthreads();
  conv_final(bufE, Wbuf + W_SA3_1, pooled, p, lane);
}

// ---- kernel 2: FC chain + axis math; one wave per 16 points -----------------
__global__ __launch_bounds__(128) void fc_kernel(
    const _Float16* __restrict__ pooled, const _Float16* __restrict__ Wbuf,
    const float* __restrict__ Bbuf, float* __restrict__ out)
{
  __shared__ __align__(16) _Float16 fcbuf[4*1024];  // per wave: 2x [16][32]
  __shared__ float axisbuf[4*128];                  // per wave: [16][8]
  int wave = threadIdx.x >> 5, lane = threadIdx.x & 31;
  int tile = blockIdx.x*4 + wave;                   // 8192 tiles of 16 points
  int n = lane & 15, kh = lane >> 4;
  _Float16* b1 = fcbuf + wave*1024;
  _Float16* b2 = b1 + 512;
  float* ab = axisbuf + wave*128;

  // fc1: 64 -> 32, relu (A read straight from global pooled[tile*16 ..][64])
  conv_layer<2,2,true>(pooled + (long)tile*16*64, b1, 32, 0, 32,
                       Wbuf + W_FC1, Bbuf + B_FC1, lane);
  __syncthreads();
  // fc2: 32 -> 32, relu
  conv_layer<1,2,true>(b1, b2, 32, 0, 32, Wbuf + W_FC2, Bbuf + B_FC2, lane);
  __syncthreads();
  // fc3: 32 -> 6 (padded 16), tanh
  {
    v16h A[1];
    loadA<1>(b2, lane, A);
    v8f acc = {};
    v16h Bf = *(const v16h*)(Wbuf + W_FC3 + n*32 + kh*16);
    acc = wmma16(A[0], Bf, acc);
    float bv = Bbuf[B_FC3 + n];
    if (n < 6) {
#pragma unroll
      for (int v = 0; v < 8; v++)
        ab[(v + 8*kh)*8 + n] = tanhf(acc[v] + bv);
    }
  }
  __syncthreads();
  // per-point Gram-Schmidt axes (lanes 0-15, one point each)
  if (lane < 16) {
    const float* q = ab + lane*8;
    float a10=q[0], a11=q[1], a12=q[2], a20=q[3], a21=q[4], a22=q[5];
    float a1n = sqrtf(a10*a10 + a11*a11 + a12*a12) + 1e-9f;
    float k   = (a10*a20 + a11*a21 + a12*a22) / (a1n*a1n);
    float b20 = a20 - k*a10, b21 = a21 - k*a11, b22 = a22 - k*a12;
    float bn  = sqrtf(b20*b20 + b21*b21 + b22*b22) + 1e-9f;
    float xx = b20/bn, xy = b21/bn, xz = b22/bn;
    float zx = a10/a1n, zy = a11/a1n, zz = a12/a1n;
    float yx = zy*xz - zz*xy, yy = zz*xx - zx*xz, yz = zx*xy - zy*xx;
    long p = (long)tile*16 + lane;
    const long BN3 = (long)NPTS*3;
    float* ox = out + p*3;        ox[0]=xx; ox[1]=xy; ox[2]=xz;
    float* oy = out + BN3 + p*3;  oy[0]=yx; oy[1]=yy; oy[2]=yz;
    float* oz = out + 2*BN3+p*3;  oz[0]=zx; oz[1]=zy; oz[2]=zz;
  }
}

// ---- launch -----------------------------------------------------------------
extern "C" void kernel_launch(void* const* d_in, const int* in_sizes, int n_in,
                              void* d_out, int out_size, void* d_ws, size_t ws_size,
                              hipStream_t stream) {
  const float* xyz = (const float*)d_in[0];
  const int*   nbr = (const int*)d_in[1];
  PrepParams P;
  for (int l = 0; l < 6; l++) {
    P.W[l]  = (const float*)d_in[2 + 4*l];
    P.b[l]  = (const float*)d_in[3 + 4*l];
    P.ga[l] = (const float*)d_in[4 + 4*l];
    P.be[l] = (const float*)d_in[5 + 4*l];
  }
  for (int l = 0; l < 3; l++) {
    P.fv[l] = (const float*)d_in[26 + 3*l];
    P.fg[l] = (const float*)d_in[27 + 3*l];
    P.fb[l] = (const float*)d_in[28 + 3*l];
  }
  _Float16* Wbuf   = (_Float16*)d_ws;
  float*    Bbuf   = (float*)((char*)d_ws + 28672);
  _Float16* pooled = (_Float16*)((char*)d_ws + 65536);

  prep_kernel<<<1, 256, 0, stream>>>(P, Wbuf, Bbuf);
  conv_kernel<<<NPTS/4, 128, 0, stream>>>(xyz, nbr, Wbuf, pooled);
  fc_kernel<<<NPTS/64, 128, 0, stream>>>(pooled, Wbuf, Bbuf, (float*)d_out);
}
